// CompressedSensingInception_14516989460534
// MI455X (gfx1250) — matrix operand order repaired and checked by reading, not compiled, not run on hardware
//
#include <hip/hip_runtime.h>
#include <hip/hip_bf16.h>
#include <cmath>
#include <cstdint>

typedef __attribute__((ext_vector_type(16))) _Float16 v16h;
typedef __attribute__((ext_vector_type(8)))  float    v8f;

// ---------------------------------------------------------------------------
// Problem constants
// ---------------------------------------------------------------------------
#define BATCH   128
#define NBC     384            // B * 3 channels
#define LDIM    5184           // fine-grid size (72*72)
#define KDIM    81             // coarse size (9*9)
#define KPAD    96             // 81 padded to 6*16
#define KSPLIT  6              // K-split for GEMM1 (5184 / 6 = 864 = 27*32)
#define KCHUNK  864
#define FISTA_ITERS 100

union FragU { v16h h; unsigned u[8]; };

__device__ inline v16h load_a16(const _Float16* __restrict__ row, int k0, int hi) {
  // 16-bit A-matrix 16x32 layout: VGPR v<4 : K = k0 + 8*hi + 2v
  //                               VGPR v>=4: K = k0 + 16 + 8*hi + 2(v-4)
  FragU f;
#pragma unroll
  for (int v = 0; v < 4; ++v)
    f.u[v] = *(const unsigned*)(row + k0 + 8 * hi + 2 * v);
#pragma unroll
  for (int v = 4; v < 8; ++v)
    f.u[v] = *(const unsigned*)(row + k0 + 16 + 8 * hi + 2 * (v - 4));
  return f.h;
}

__device__ inline v16h load_b16(const _Float16* __restrict__ row, int k0, int hi) {
  // 16-bit B-matrix 32x16 layout (row = N-major line): VGPR v<4 : K = k0 + 8*hi + 2v
  //                                                   VGPR v>=4: K = k0 + 16 + 8*hi + 2(v-4)
  FragU f;
#pragma unroll
  for (int v = 0; v < 8; ++v)
    f.u[v] = *(const unsigned*)(row + k0 + ((v >> 2) << 4) + 8 * hi + 2 * (v & 3));
  return f.h;
}

__device__ inline float lrelu(float x) { return x > 0.f ? x : 0.3f * x; }

// ---------------------------------------------------------------------------
// Init: zero FISTA state (harness poisons ws; y0 = 0 every call)
// ---------------------------------------------------------------------------
__global__ void k_zero(float* __restrict__ Ymom, float* __restrict__ Ylast,
                       _Float16* __restrict__ Yh) {
  size_t i = (size_t)blockIdx.x * blockDim.x + threadIdx.x;
  const size_t N = (size_t)NBC * LDIM;
  if (i < N) { Ymom[i] = 0.f; Ylast[i] = 0.f; Yh[i] = (_Float16)0.f; }
}

// Convert PSF matrix to f16 in the two layouts the GEMMs want.
__global__ void k_matcvt(const float* __restrict__ mat,
                         _Float16* __restrict__ mat_h,     // (5184 x 96) K-contig
                         _Float16* __restrict__ matT_h) {  // (96 x 5184) L-contig
  int i = blockIdx.x * blockDim.x + threadIdx.x;
  if (i >= LDIM * KPAD) return;
  int l = i / KPAD, k = i % KPAD;
  float v = (k < KDIM) ? mat[(size_t)l * KDIM + k] : 0.f;
  mat_h[i] = (_Float16)v;
  matT_h[(size_t)k * LDIM + l] = (_Float16)v;
}

// ---------------------------------------------------------------------------
// z-path: BN -> conv5x5 s3 -> BN lrelu -> conv5x5 s3 -> BN lrelu -> 3 dense
//         -> lambda = 0.1 * 0.3 * sigmoid(.)   (one block of 32 per batch)
// ---------------------------------------------------------------------------
__global__ void k_zpath(const float* __restrict__ in,
                        const float* bn0_g, const float* bn0_b, const float* bn0_m, const float* bn0_v,
                        const float* d1_k, const float* d1_g, const float* d1_b, const float* d1_m, const float* d1_v,
                        const float* d2_k, const float* d2_g, const float* d2_b, const float* d2_m, const float* d2_v,
                        const float* h1_w, const float* h1_b, const float* h2_w, const float* h2_b,
                        const float* h3_w, const float* h3_b, float* __restrict__ lam) {
  __shared__ float s1[108];
  __shared__ float s2[24];
  __shared__ float s3[24];
  __shared__ float s4[12];
  const int b = blockIdx.x, tid = threadIdx.x;
  float sc0[3], sh0[3];
  for (int c = 0; c < 3; ++c) {
    float s = bn0_g[c] * rsqrtf(bn0_v[c] + 1e-3f);
    sc0[c] = s; sh0[c] = bn0_b[c] - bn0_m[c] * s;
  }
  const float* ib = in + (size_t)b * 9 * 9 * 3;
  for (int idx = tid; idx < 108; idx += 32) {
    int co = idx % 12, ow = (idx / 12) % 3, oh = idx / 36;
    float acc = 0.f;
    for (int kh = 0; kh < 5; ++kh) {
      int ih = oh * 3 - 1 + kh; if (ih < 0 || ih >= 9) continue;
      for (int kw = 0; kw < 5; ++kw) {
        int iw = ow * 3 - 1 + kw; if (iw < 0 || iw >= 9) continue;
        for (int ci = 0; ci < 3; ++ci)
          acc += (ib[(ih * 9 + iw) * 3 + ci] * sc0[ci] + sh0[ci]) * d1_k[((kh * 5 + kw) * 3 + ci) * 12 + co];
      }
    }
    float s = d1_g[co] * rsqrtf(d1_v[co] + 1e-3f);
    s1[idx] = lrelu((acc - d1_m[co]) * s + d1_b[co]);
  }
  __syncthreads();
  if (tid < 24) {
    float acc = 0.f;
    for (int kh = 0; kh < 5; ++kh) {
      int ih = -1 + kh; if (ih < 0 || ih >= 3) continue;
      for (int kw = 0; kw < 5; ++kw) {
        int iw = -1 + kw; if (iw < 0 || iw >= 3) continue;
        for (int ci = 0; ci < 12; ++ci)
          acc += s1[(ih * 3 + iw) * 12 + ci] * d2_k[((kh * 5 + kw) * 12 + ci) * 24 + tid];
      }
    }
    float s = d2_g[tid] * rsqrtf(d2_v[tid] + 1e-3f);
    s2[tid] = lrelu((acc - d2_m[tid]) * s + d2_b[tid]);
  }
  __syncthreads();
  if (tid < 24) { float a = h1_b[tid]; for (int i = 0; i < 24; ++i) a += s2[i] * h1_w[i * 24 + tid]; s3[tid] = a; }
  __syncthreads();
  if (tid < 12) { float a = h2_b[tid]; for (int i = 0; i < 24; ++i) a += s3[i] * h2_w[i * 12 + tid]; s4[tid] = a; }
  __syncthreads();
  if (tid == 0) {
    float a = h3_b[0];
    for (int i = 0; i < 12; ++i) a += s4[i] * h3_w[i];
    float p = 0.3f / (1.f + __expf(-a));
    float lv = 0.1f * p;
    lam[b * 3 + 0] = lv; lam[b * 3 + 1] = lv; lam[b * 3 + 2] = lv;
  }
}

// ---------------------------------------------------------------------------
// x-path head: conv1x1 + lrelu + per-(b,c) max-abs normalize -> IM (bc x 96)
// one block (96 threads) per (b,c)
// ---------------------------------------------------------------------------
__global__ void k_xpath(const float* __restrict__ in, const float* cx1_k, const float* cx1_b,
                        float* __restrict__ IMf) {
  const int bc = blockIdx.x, b = bc / 3, c = bc % 3;
  const int tid = threadIdx.x;
  __shared__ float sx[81];
  __shared__ float smax;
  if (tid < 81) {
    const float* ib = in + ((size_t)b * 81 + tid) * 3;
    float a = cx1_b[c];
    for (int ci = 0; ci < 3; ++ci) a += ib[ci] * cx1_k[ci * 3 + c];
    sx[tid] = lrelu(a);
  }
  __syncthreads();
  if (tid == 0) {
    float m = 0.f;
    for (int i = 0; i < 81; ++i) m = fmaxf(m, fabsf(sx[i]));
    smax = m;
  }
  __syncthreads();
  float inv = 1.f / (0.001f + smax);
  IMf[(size_t)bc * KPAD + tid] = (tid < 81) ? sx[tid] * inv : 0.f;
}

// ---------------------------------------------------------------------------
// FISTA GEMM1 (K-split): Rp[kc] = matT(96x5184) * Y(5184x384)   [f16 WMMA]
// One wave per 16x16 tile per K-chunk; 864 waves.
// The 4 waves of a block share the same (kc, tm) A-chunk (4|24, 4|144), so the
// 16x864 f16 A slab is staged once per block via async global->LDS copies
// (ASYNCcnt). The B stream is ping-pong software-pipelined against the WMMAs:
// 12 branch-free double-steps + 3-step epilogue = 27 k-steps, no register
// copies, no per-iteration guard.
// ---------------------------------------------------------------------------
__global__ void __launch_bounds__(128) k_fista1(const _Float16* __restrict__ matT_h,
                                                const _Float16* __restrict__ Yh,
                                                float* __restrict__ Rp) {
  __shared__ _Float16 sA[16 * KCHUNK];             // 27648 B per block
  const int tid  = threadIdx.x;
  const int wave = tid >> 5;
  const int lane = tid & 31;
  const int tile = blockIdx.x * 4 + wave;          // 0 .. 863
  const int kc = tile / 144;
  const int r  = tile % 144;
  const int tm = r / 24, tn = r % 24;
  const int hi = lane >> 4, lr = lane & 15;
  const int kbeg = kc * KCHUNK;

  // ---- async-stage shared A chunk: rows tm*16..+15, cols [kbeg, kbeg+864) --
  {
    const unsigned ldsbase = (unsigned)(uintptr_t)(&sA[0]);
    const _Float16* gA = matT_h + (size_t)tm * 16 * LDIM + kbeg;
    for (int i = tid; i < 16 * KCHUNK / 8; i += 128) {      // 16-byte chunks
      int row = i / (KCHUNK / 8), c = i % (KCHUNK / 8);
      unsigned long long ga =
          (unsigned long long)(uintptr_t)(gA + (size_t)row * LDIM + c * 8);
      unsigned lo = ldsbase + (unsigned)(row * KCHUNK + c * 8) * 2;
      asm volatile("global_load_async_to_lds_b128 %0, %1, off"
                   :: "v"(lo), "v"(ga) : "memory");
    }
    asm volatile("s_wait_asynccnt 0x0" ::: "memory");
  }
  __syncthreads();

  const _Float16* As   = &sA[lr * KCHUNK];                  // LDS, row lr
  const _Float16* Brow = Yh + (size_t)(tn * 16 + lr) * LDIM;

  v8f acc = {};
  // pipeline prologue: two B fragments in flight
  v16h b0 = load_b16(Brow, kbeg, hi);
  v16h b1 = load_b16(Brow, kbeg + 32, hi);
  int off = 0;                                              // k - kbeg
#pragma unroll 4
  for (int s = 0; s < 12; ++s) {                            // 24 k-steps
    v16h a0 = load_a16(As, off, hi);
    acc = __builtin_amdgcn_wmma_f32_16x16x32_f16(false, a0, false, b0, (short)0, acc, false, false);
    b0 = load_b16(Brow, kbeg + off + 64, hi);
    __builtin_prefetch(Brow + kbeg + off + 192, 0, 2);
    v16h a1 = load_a16(As, off + 32, hi);
    acc = __builtin_amdgcn_wmma_f32_16x16x32_f16(false, a1, false, b1, (short)0, acc, false, false);
    b1 = load_b16(Brow, kbeg + off + 96, hi);
    off += 64;
  }
  // epilogue: off == 768; b0 = B(768), b1 = B(800); 3 k-steps remain
  {
    v16h b2 = load_b16(Brow, kbeg + 832, hi);
    v16h a  = load_a16(As, 768, hi);
    acc = __builtin_amdgcn_wmma_f32_16x16x32_f16(false, a, false, b0, (short)0, acc, false, false);
    a = load_a16(As, 800, hi);
    acc = __builtin_amdgcn_wmma_f32_16x16x32_f16(false, a, false, b1, (short)0, acc, false, false);
    a = load_a16(As, 832, hi);
    acc = __builtin_amdgcn_wmma_f32_16x16x32_f16(false, a, false, b2, (short)0, acc, false, false);
  }

  const int bc = tn * 16 + lr;
  const int kb = tm * 16 + 8 * hi;
  float* out = Rp + ((size_t)kc * NBC + bc) * KPAD + kb;
#pragma unroll
  for (int v = 0; v < 8; ++v) out[v] = acc[v];
}

// Reduce K-split partials, subtract from IM, convert to f16 R (N-major).
__global__ void k_r2h(const float* __restrict__ IMf, const float* __restrict__ Rp,
                      _Float16* __restrict__ Rh) {
  int i = blockIdx.x * blockDim.x + threadIdx.x;
  if (i >= NBC * KPAD) return;
  float s = 0.f;
#pragma unroll
  for (int kc = 0; kc < KSPLIT; ++kc) s += Rp[(size_t)kc * NBC * KPAD + i];
  Rh[i] = (_Float16)(IMf[i] - s);
}

// ---------------------------------------------------------------------------
// FISTA GEMM2 + fused update: U = mat(5184x96) * R(96x384)
//   y_new = soft(y_tmp + U, lam); y_mom = y_new + beta*(y_new - y_last)
// 7776 waves (16x16 tiles), K = 96 (3 WMMA steps)
// ---------------------------------------------------------------------------
__global__ void k_fista2(const _Float16* __restrict__ mat_h, const _Float16* __restrict__ Rh,
                         const float* __restrict__ lam, float beta,
                         float* __restrict__ Ymom, float* __restrict__ Ylast,
                         _Float16* __restrict__ Yh) {
  const int wave = threadIdx.x >> 5;
  const int lane = threadIdx.x & 31;
  const int tile = blockIdx.x * 8 + wave;          // 0 .. 7775
  const int tm = tile / 24, tn = tile % 24;
  const int hi = lane >> 4, lr = lane & 15;
  const _Float16* Arow = mat_h + (size_t)(tm * 16 + lr) * KPAD;
  const _Float16* Brow = Rh    + (size_t)(tn * 16 + lr) * KPAD;
  v8f acc = {};
#pragma unroll
  for (int k0 = 0; k0 < KPAD; k0 += 32) {
    v16h a = load_a16(Arow, k0, hi);
    v16h bb = load_b16(Brow, k0, hi);
    acc = __builtin_amdgcn_wmma_f32_16x16x32_f16(false, a, false, bb, (short)0, acc, false, false);
  }
  const int bc = tn * 16 + lr;
  const float lamv = lam[bc];
  const size_t base = (size_t)bc * LDIM + tm * 16 + 8 * hi;
#pragma unroll
  for (int v = 0; v < 8; ++v) {
    size_t idx = base + v;
    float val  = Ymom[idx] + acc[v];
    float ynew = fmaxf(val - lamv, 0.f) - fmaxf(-val - lamv, 0.f);
    float ymo  = ynew + beta * (ynew - Ylast[idx]);
    Ylast[idx] = ynew;
    Ymom[idx]  = ymo;
    Yh[idx]    = (_Float16)ymo;
  }
}

// ---------------------------------------------------------------------------
// Post: cs_out, w/y paths, conv stack
// ---------------------------------------------------------------------------
__global__ void k_csout(const float* __restrict__ Ylast, float* __restrict__ out2) {
  size_t i = (size_t)blockIdx.x * blockDim.x + threadIdx.x;
  const size_t N = (size_t)BATCH * 72 * 72 * 3;
  if (i >= N) return;
  int c = i % 3; size_t p = i / 3;
  int w = p % 72; size_t q = p / 72;
  int h = q % 72; int b = (int)(q / 72);
  out2[i] = Ylast[(size_t)(b * 3 + c) * LDIM + h * 72 + w];
}

__global__ void k_wy(const float* __restrict__ in, const float* w1_k, const float* w1_b,
                     const float* y17_k, const float* y17_b,
                     const float* y71_k, const float* y71_b,
                     const float* y11_k, const float* y11_b, float* __restrict__ out) {
  const int bp = blockIdx.x;            // b*81 + h*9 + w
  const int b = bp / 81, p = bp % 81, h = p / 9, w = p % 9;
  const int tid = threadIdx.x;          // 32
  __shared__ float cat[64];
  const float* ib = in + (size_t)b * 243;
  {
    float a = y17_b[tid];
    for (int kw = 0; kw < 7; ++kw) {
      int iw = w - 3 + kw; if (iw < 0 || iw >= 9) continue;
      for (int ci = 0; ci < 3; ++ci) a += ib[(h * 9 + iw) * 3 + ci] * y17_k[(kw * 3 + ci) * 32 + tid];
    }
    cat[tid] = a;
  }
  {
    float a = y71_b[tid];
    for (int kh = 0; kh < 7; ++kh) {
      int ih = h - 3 + kh; if (ih < 0 || ih >= 9) continue;
      for (int ci = 0; ci < 3; ++ci) a += ib[(ih * 9 + w) * 3 + ci] * y71_k[(kh * 3 + ci) * 32 + tid];
    }
    cat[32 + tid] = a;
  }
  __syncthreads();
  float a = y11_b[tid];
  for (int i = 0; i < 64; ++i) a += cat[i] * y11_k[i * 32 + tid];
  float* ob = out + (size_t)bp * 41;
  ob[9 + tid] = lrelu(a);
  if (tid == 0) {
    float wv = w1_b[0];
    for (int ci = 0; ci < 3; ++ci) wv += ib[p * 3 + ci] * w1_k[ci];
    ob[0] = lrelu(wv);
  }
}

__global__ void k_c51(const float* __restrict__ Ylast, const float* kk, const float* kb,
                      float* __restrict__ xp1) {
  const int id = blockIdx.x * 4 + (threadIdx.x >> 4);   // b*36*36 + oh*36 + ow
  const int co = threadIdx.x & 15;
  const int ow = id % 36, t = id / 36, oh = t % 36, b = t / 36;
  const int iw = ow * 2;
  float a = kb[co];
  for (int kh = 0; kh < 5; ++kh) {
    int ih = oh * 2 - 1 + kh; if (ih < 0 || ih >= 72) continue;
    for (int ci = 0; ci < 3; ++ci)
      a += Ylast[(size_t)(b * 3 + ci) * LDIM + ih * 72 + iw] * kk[(kh * 3 + ci) * 16 + co];
  }
  xp1[(size_t)id * 16 + co] = a;
}

__global__ void k_c15(const float* __restrict__ xp1, const float* kk, const float* kb,
                      float* __restrict__ xp2) {
  const int id = blockIdx.x;            // b*18*18 + oh*18 + ow
  const int co = threadIdx.x;           // 32
  const int ow = id % 18, t = id / 18, oh = t % 18, b = t / 18;
  const int ih = oh * 2;
  float a = kb[co];
  for (int kw = 0; kw < 5; ++kw) {
    int iw = ow * 2 - 1 + kw; if (iw < 0 || iw >= 36) continue;
    const float* xr = xp1 + (((size_t)b * 36 + ih) * 36 + iw) * 16;
    for (int ci = 0; ci < 16; ++ci) a += xr[ci] * kk[(kw * 16 + ci) * 32 + co];
  }
  xp2[(size_t)id * 32 + co] = a;
}

__global__ void k_c55(const float* __restrict__ xp2, const float* kk, const float* kb,
                      float* __restrict__ xp3) {
  const int id = blockIdx.x;            // b*81 + oh*9 + ow
  const int co = threadIdx.x;           // 64
  const int ow = id % 9, t = id / 9, oh = t % 9, b = t / 9;
  float a = kb[co];
  for (int kh = 0; kh < 5; ++kh) {
    int ih = oh * 2 - 1 + kh; if (ih < 0 || ih >= 18) continue;
    for (int kw = 0; kw < 5; ++kw) {
      int iw = ow * 2 - 1 + kw; if (iw < 0 || iw >= 18) continue;
      const float* xr = xp2 + (((size_t)b * 18 + ih) * 18 + iw) * 32;
      for (int ci = 0; ci < 32; ++ci) a += xr[ci] * kk[((kh * 5 + kw) * 32 + ci) * 64 + co];
    }
  }
  xp3[(size_t)id * 64 + co] = a;
}

__global__ void k_cx2(const float* __restrict__ xp3, const float* kk, const float* kb,
                      float* __restrict__ out) {
  const int id = blockIdx.x * 8 + (threadIdx.x >> 3);   // b*81 + p
  const int o = threadIdx.x & 7;
  float a = kb[o];
  const float* xb = xp3 + (size_t)id * 64;
  for (int i = 0; i < 64; ++i) a += xb[i] * kk[i * 8 + o];
  out[(size_t)id * 41 + 1 + o] = lrelu(a);
}

// ---------------------------------------------------------------------------
// Host launcher
// ---------------------------------------------------------------------------
extern "C" void kernel_launch(void* const* d_in, const int* in_sizes, int n_in,
                              void* d_out, int out_size, void* d_ws, size_t ws_size,
                              hipStream_t stream) {
  const float* in    = (const float*)d_in[0];
  const float* mat   = (const float*)d_in[1];
  const float* w1_k  = (const float*)d_in[2];
  const float* w1_b  = (const float*)d_in[3];
  const float* bn0_g = (const float*)d_in[4];
  const float* bn0_b = (const float*)d_in[5];
  const float* bn0_m = (const float*)d_in[6];
  const float* bn0_v = (const float*)d_in[7];
  const float* d1_k  = (const float*)d_in[8];
  const float* d1_g  = (const float*)d_in[9];
  const float* d1_b  = (const float*)d_in[10];
  const float* d1_m  = (const float*)d_in[11];
  const float* d1_v  = (const float*)d_in[12];
  const float* d2_k  = (const float*)d_in[13];
  const float* d2_g  = (const float*)d_in[14];
  const float* d2_b  = (const float*)d_in[15];
  const float* d2_m  = (const float*)d_in[16];
  const float* d2_v  = (const float*)d_in[17];
  const float* h1_w  = (const float*)d_in[18];
  const float* h1_b  = (const float*)d_in[19];
  const float* h2_w  = (const float*)d_in[20];
  const float* h2_b  = (const float*)d_in[21];
  const float* h3_w  = (const float*)d_in[22];
  const float* h3_b  = (const float*)d_in[23];
  const float* cx1_k = (const float*)d_in[24];
  const float* cx1_b = (const float*)d_in[25];
  const float* c51_k = (const float*)d_in[26];
  const float* c51_b = (const float*)d_in[27];
  const float* c15_k = (const float*)d_in[28];
  const float* c15_b = (const float*)d_in[29];
  const float* c55_k = (const float*)d_in[30];
  const float* c55_b = (const float*)d_in[31];
  const float* cx2_k = (const float*)d_in[32];
  const float* cx2_b = (const float*)d_in[33];
  const float* y17_k = (const float*)d_in[34];
  const float* y17_b = (const float*)d_in[35];
  const float* y71_k = (const float*)d_in[36];
  const float* y71_b = (const float*)d_in[37];
  const float* y11_k = (const float*)d_in[38];
  const float* y11_b = (const float*)d_in[39];

  // ---- workspace carve-up (deterministic) --------------------------------
  char* ws = (char*)d_ws;
  size_t off = 0;
  auto carve = [&](size_t bytes) { size_t r = off; off += (bytes + 255) & ~(size_t)255; return r; };
  float*    lam    = (float*)(ws + carve((size_t)NBC * 4));
  float*    IMf    = (float*)(ws + carve((size_t)NBC * KPAD * 4));
  _Float16* matT_h = (_Float16*)(ws + carve((size_t)KPAD * LDIM * 2));
  _Float16* mat_h  = (_Float16*)(ws + carve((size_t)LDIM * KPAD * 2));
  _Float16* Rh     = (_Float16*)(ws + carve((size_t)NBC * KPAD * 2));
  float*    Rp     = (float*)(ws + carve((size_t)KSPLIT * NBC * KPAD * 4));
  float*    Ylast  = (float*)(ws + carve((size_t)NBC * LDIM * 4));
  size_t regionA   = carve((size_t)NBC * LDIM * 4 + (size_t)NBC * LDIM * 2);
  float*    Ymom   = (float*)(ws + regionA);
  _Float16* Yh     = (_Float16*)(ws + regionA + (size_t)NBC * LDIM * 4);
  // post-FISTA aliases (dead state reused as conv scratch)
  float* xp1 = (float*)(ws + regionA);   // 128*36*36*16 f32 = 10.6MB <= 11.9MB
  float* xp2 = Ylast;                    // 128*18*18*32 f32 = 5.3MB  <= 7.96MB (Ylast dead after c51)
  float* xp3 = (float*)(ws + regionA);   // 128*9*9*64  f32 = 2.65MB (xp1 dead after c15)

  float* out  = (float*)d_out;                       // (128,9,9,41)
  float* cs   = out + (size_t)BATCH * 81 * 41;       // (128,72,72,3)

  // ---- prep ---------------------------------------------------------------
  k_zero  <<<7776, 256, 0, stream>>>(Ymom, Ylast, Yh);
  k_matcvt<<<(LDIM * KPAD + 255) / 256, 256, 0, stream>>>(mat, mat_h, matT_h);
  k_zpath <<<BATCH, 32, 0, stream>>>(in, bn0_g, bn0_b, bn0_m, bn0_v,
                                     d1_k, d1_g, d1_b, d1_m, d1_v,
                                     d2_k, d2_g, d2_b, d2_m, d2_v,
                                     h1_w, h1_b, h2_w, h2_b, h3_w, h3_b, lam);
  k_xpath <<<NBC, 96, 0, stream>>>(in, cx1_k, cx1_b, IMf);

  // ---- FISTA loop (deterministic momentum schedule) -----------------------
  float t = 1.f;
  for (int it = 0; it < FISTA_ITERS; ++it) {
    float tn   = 0.5f * (1.f + sqrtf(1.f + 4.f * t * t));
    float beta = (t - 1.f) / tn;
    t = tn;
    k_fista1<<<216, 128, 0, stream>>>(matT_h, Yh, Rp);                 // 864 waves
    k_r2h   <<<(NBC * KPAD + 255) / 256, 256, 0, stream>>>(IMf, Rp, Rh);
    k_fista2<<<972, 256, 0, stream>>>(mat_h, Rh, lam, beta, Ymom, Ylast, Yh); // 7776 waves
  }

  // ---- post ---------------------------------------------------------------
  k_csout<<<(BATCH * 72 * 72 * 3 + 255) / 256, 256, 0, stream>>>(Ylast, cs);
  k_wy   <<<BATCH * 81, 32, 0, stream>>>(in, w1_k, w1_b, y17_k, y17_b,
                                         y71_k, y71_b, y11_k, y11_b, out);
  k_c51  <<<BATCH * 36 * 36 / 4, 64, 0, stream>>>(Ylast, c51_k, c51_b, xp1);
  k_c15  <<<BATCH * 18 * 18, 32, 0, stream>>>(xp1, c15_k, c15_b, xp2);
  k_c55  <<<BATCH * 81, 64, 0, stream>>>(xp2, c55_k, c55_b, xp3);
  k_cx2  <<<BATCH * 81 / 8, 64, 0, stream>>>(xp3, cx2_k, cx2_b, out);
}
